// AttentionHead_4741643895124
// MI455X (gfx1250) — compile-verified, matrix operand order checked by reading, not run on hardware
//
#include <hip/hip_runtime.h>
#include <hip/hip_bf16.h>

// ---- types ----
typedef _Float16 h16v __attribute__((ext_vector_type(16)));
typedef _Float16 h8v  __attribute__((ext_vector_type(8)));
typedef float    f8v  __attribute__((ext_vector_type(8)));
typedef float    f4v  __attribute__((ext_vector_type(4)));

#define D_MODEL 1024
#define HEAD    64
#define SEQ     4096
#define BATCH   4
#define NROWS   (BATCH * SEQ)   // 16384 flat rows

__device__ __forceinline__ h16v hcat(h8v lo, h8v hi) {
  h16v r;
#pragma unroll
  for (int i = 0; i < 8; ++i) { r[i] = lo[i]; r[i + 8] = hi[i]; }
  return r;
}

__device__ __forceinline__ float redmax16(float v) {
#pragma unroll
  for (int off = 1; off < 16; off <<= 1) v = fmaxf(v, __shfl_xor(v, off, 32));
  return v;
}
__device__ __forceinline__ float redsum16(float v) {
#pragma unroll
  for (int off = 1; off < 16; off <<= 1) v += __shfl_xor(v, off, 32);
  return v;
}

// LDS byte offset of a __shared__ object: generic addr low 32 bits == LDS addr.
__device__ __forceinline__ unsigned lds_off(const void* p) {
  return (unsigned)(size_t)p;
}

// Per-lane async copy: 16 bytes global -> LDS, tracked by ASYNCcnt.
__device__ __forceinline__ void async_b128(unsigned lds_byte_off, const void* gaddr) {
  asm volatile("global_load_async_to_lds_b128 %0, %1, off"
               :: "v"(lds_byte_off), "v"(gaddr) : "memory");
}

// ---------------------------------------------------------------------------
// Kernel 0: WhT[m][n][k] = (f16) W_m[k][n]   (3 x 64 x 1024 halves = 384 KB)
// ---------------------------------------------------------------------------
__global__ __launch_bounds__(256) void wtrans_kernel(
    const float* __restrict__ Wq, const float* __restrict__ Wk,
    const float* __restrict__ Wv, _Float16* __restrict__ WhT) {
  int idx = blockIdx.x * 256 + threadIdx.x;   // < 3*64*1024
  int m = idx >> 16;
  int r = idx & 65535;
  int n = r >> 10;
  int k = r & 1023;
  const float* W = (m == 0) ? Wq : ((m == 1) ? Wk : Wv);
  WhT[idx] = (_Float16)W[k * HEAD + n];
}

// ---------------------------------------------------------------------------
// Kernel 1: fused Q/K/V projection.  4 waves/block, 64 rows/block, all three
// matrices per block.  W tiles (3 x [64n x 32k] f16 = 12 KB) double-buffered
// in LDS via async global->LDS copies; 12 WMMAs per k-step per wave.
//   Q -> Qh  [row][h] f16, (x*Wq+bq)*0.125 ;  K -> Kh [row][h] ; V -> VhT[b][h][kv]
// ---------------------------------------------------------------------------
__global__ __launch_bounds__(128) void proj_kernel(
    const float* __restrict__ x, const _Float16* __restrict__ WhT,
    const float* __restrict__ bq, const float* __restrict__ bk,
    const float* __restrict__ bv, _Float16* __restrict__ Qh,
    _Float16* __restrict__ Kh, _Float16* __restrict__ VhT) {
  __shared__ __align__(16) _Float16 Wbuf[2][3 * 64 * 32];   // 2 x 12 KB

  const int tid    = threadIdx.x;
  const int wave   = tid >> 5;
  const int lane   = tid & 31;
  const int half16 = lane >> 4;
  const int l16    = lane & 15;
  const int hb     = half16 * 8;
  const int row0   = blockIdx.x * 64 + wave * 16;      // this wave's 16-row tile
  const float* xrow = x + (size_t)(row0 + l16) * D_MODEL;

  f8v acc[3][4] = {};

  // prefetch W tile for kk = 0 into buffer 0
  {
#pragma unroll
    for (int i = 0; i < 6; ++i) {
      const int c  = tid + 128 * i;        // 0..767 chunks of 16B (12 KB)
      const int m  = c >> 8, r = c & 255;
      const int n  = r >> 2, ko = (r & 3) * 8;
      async_b128(lds_off(&Wbuf[0][m * 2048 + n * 32 + ko]),
                 WhT + (size_t)m * 65536 + (size_t)n * 1024 + ko);
    }
  }

  for (int kk = 0; kk < D_MODEL; kk += 32) {
    const int cur = (kk >> 5) & 1;
    if (kk + 32 < D_MODEL) {
#pragma unroll
      for (int i = 0; i < 6; ++i) {
        const int c  = tid + 128 * i;
        const int m  = c >> 8, r = c & 255;
        const int n  = r >> 2, ko = (r & 3) * 8;
        async_b128(lds_off(&Wbuf[cur ^ 1][m * 2048 + n * 32 + ko]),
                   WhT + (size_t)m * 65536 + (size_t)n * 1024 + (kk + 32) + ko);
      }
      asm volatile("s_wait_asynccnt 0x6" ::: "memory");   // current buffer done
    } else {
      asm volatile("s_wait_asynccnt 0x0" ::: "memory");
    }
    __syncthreads();

    // A tile from global: 16 rows x 32 k of x, fp32 -> f16
    const int kA = kk + hb;
    f4v a0 = *(const f4v*)(xrow + kA);
    f4v a1 = *(const f4v*)(xrow + kA + 4);
    f4v a2 = *(const f4v*)(xrow + kA + 16);
    f4v a3 = *(const f4v*)(xrow + kA + 20);
    h16v A;
#pragma unroll
    for (int i = 0; i < 4; ++i) {
      A[i]      = (_Float16)a0[i];
      A[4 + i]  = (_Float16)a1[i];
      A[8 + i]  = (_Float16)a2[i];
      A[12 + i] = (_Float16)a3[i];
    }

    const _Float16* wb = &Wbuf[cur][0];
#pragma unroll
    for (int m = 0; m < 3; ++m) {
#pragma unroll
      for (int t = 0; t < 4; ++t) {
        const _Float16* wcol = wb + m * 2048 + (t * 16 + l16) * 32;
        h16v B = hcat(*(const h8v*)(wcol + hb), *(const h8v*)(wcol + hb + 16));
        acc[m][t] = __builtin_amdgcn_wmma_f32_16x16x32_f16(
            false, A, false, B, (short)0, acc[m][t], false, false);
      }
    }
    __syncthreads();   // don't let next prefetch overwrite a buffer in use
  }

  // bias + scale + store
#pragma unroll
  for (int m = 0; m < 3; ++m) {
    const float* bias = (m == 0) ? bq : ((m == 1) ? bk : bv);
    const float  scl  = (m == 0) ? 0.125f : 1.0f;
#pragma unroll
    for (int t = 0; t < 4; ++t) {
      const int   col = t * 16 + l16;
      const float bvv = bias[col];
#pragma unroll
      for (int g = 0; g < 8; ++g) {
        const int      r    = row0 + g + 8 * half16;
        const _Float16 hval = (_Float16)((acc[m][t][g] + bvv) * scl);
        if (m == 0) {
          Qh[(size_t)r * HEAD + col] = hval;
        } else if (m == 1) {
          Kh[(size_t)r * HEAD + col] = hval;
        } else {
          const int bb = r >> 12, kv = r & (SEQ - 1);
          VhT[((size_t)bb * HEAD + col) * SEQ + kv] = hval;
        }
      }
    }
  }
}

// ---------------------------------------------------------------------------
// Kernel 2: flash attention.  4 waves/block sharing double-buffered K/V tiles
// in LDS (async global->LDS).  Each wave: 16 q-rows, online softmax, kv step 32.
// ---------------------------------------------------------------------------
__global__ __launch_bounds__(128) void attn_kernel(
    const _Float16* __restrict__ Qh, const _Float16* __restrict__ Kh,
    const _Float16* __restrict__ VhT, float* __restrict__ out) {
  __shared__ __align__(16) _Float16 Kbuf[2][32 * 64];   // [kv][h], 2 x 4 KB
  __shared__ __align__(16) _Float16 Vbuf[2][64 * 32];   // [h][kv], 2 x 4 KB
  __shared__ __align__(16) _Float16 pbuf[4][16 * 32];   // per-wave P staging

  const int tid    = threadIdx.x;
  const int wave   = tid >> 5;
  const int lane   = tid & 31;
  const int half16 = lane >> 4;
  const int l16    = lane & 15;
  const int hb     = half16 * 8;
  const int q0     = blockIdx.x * 64 + wave * 16;   // flat q row base
  const int b      = q0 >> 12;

  const _Float16* Kh_b  = Kh  + (size_t)b * SEQ * HEAD;
  const _Float16* VhT_b = VhT + (size_t)b * HEAD * SEQ;
  _Float16* pb = pbuf[wave];

  // Q A-tiles held in VGPRs for the whole loop
  const _Float16* qrow = Qh + (size_t)(q0 + l16) * HEAD;
  const h16v aQ0 = hcat(*(const h8v*)(qrow + hb),      *(const h8v*)(qrow + hb + 16));
  const h16v aQ1 = hcat(*(const h8v*)(qrow + 32 + hb), *(const h8v*)(qrow + 32 + hb + 16));

  float mrow[8], lrow[8];
  f8v   o[4] = {};
#pragma unroll
  for (int g = 0; g < 8; ++g) { mrow[g] = -INFINITY; lrow[g] = 0.0f; }

  // prefetch kv tile 0 into buffer 0 (K: 4 KB contiguous; V: 64 rows x 64 B)
#pragma unroll
  for (int i = 0; i < 2; ++i) {
    const int c = tid + 128 * i;                       // 0..255
    async_b128(lds_off(&Kbuf[0][0]) + c * 16,
               (const char*)(Kh_b) + c * 16);
    const int vr = c >> 2, vo = (c & 3) * 8;
    async_b128(lds_off(&Vbuf[0][vr * 32 + vo]),
               VhT_b + (size_t)vr * SEQ + vo);
  }

  for (int kv = 0; kv < SEQ; kv += 32) {
    const int cur = (kv >> 5) & 1;
    if (kv + 32 < SEQ) {
      const int nkv = kv + 32;
#pragma unroll
      for (int i = 0; i < 2; ++i) {
        const int c = tid + 128 * i;
        async_b128(lds_off(&Kbuf[cur ^ 1][0]) + c * 16,
                   (const char*)(Kh_b + (size_t)nkv * HEAD) + c * 16);
        const int vr = c >> 2, vo = (c & 3) * 8;
        async_b128(lds_off(&Vbuf[cur ^ 1][vr * 32 + vo]),
                   VhT_b + (size_t)vr * SEQ + nkv + vo);
      }
      asm volatile("s_wait_asynccnt 0x4" ::: "memory");
    } else {
      asm volatile("s_wait_asynccnt 0x0" ::: "memory");
    }
    __syncthreads();

    // ---- scores: two 16x16 tiles from LDS K ----
    f8v s0, s1;
    {
      const _Float16* krow = &Kbuf[cur][l16 * 64];            // col = l16
      h16v bK0 = hcat(*(const h8v*)(krow + hb),      *(const h8v*)(krow + hb + 16));
      h16v bK1 = hcat(*(const h8v*)(krow + 32 + hb), *(const h8v*)(krow + 32 + hb + 16));
      f8v  z   = {};
      s0 = __builtin_amdgcn_wmma_f32_16x16x32_f16(false, aQ0, false, bK0, (short)0, z,  false, false);
      s0 = __builtin_amdgcn_wmma_f32_16x16x32_f16(false, aQ1, false, bK1, (short)0, s0, false, false);
      const _Float16* krow1 = &Kbuf[cur][(16 + l16) * 64];    // col = 16 + l16
      h16v cK0 = hcat(*(const h8v*)(krow1 + hb),      *(const h8v*)(krow1 + hb + 16));
      h16v cK1 = hcat(*(const h8v*)(krow1 + 32 + hb), *(const h8v*)(krow1 + 32 + hb + 16));
      s1 = __builtin_amdgcn_wmma_f32_16x16x32_f16(false, aQ0, false, cK0, (short)0, z,  false, false);
      s1 = __builtin_amdgcn_wmma_f32_16x16x32_f16(false, aQ1, false, cK1, (short)0, s1, false, false);
    }

    // ---- online softmax over these 32 columns ----
#pragma unroll
    for (int g = 0; g < 8; ++g) {
      const float tm   = fmaxf(redmax16(s0[g]), redmax16(s1[g]));
      const float mn   = fmaxf(mrow[g], tm);
      const float corr = __expf(mrow[g] - mn);
      mrow[g] = mn;
      const float p0 = __expf(s0[g] - mn);
      const float p1 = __expf(s1[g] - mn);
      lrow[g] = lrow[g] * corr + redsum16(p0 + p1);
#pragma unroll
      for (int t = 0; t < 4; ++t) o[t][g] *= corr;
      const int M = g + hb;
      pb[M * 32 + l16]      = (_Float16)p0;
      pb[M * 32 + 16 + l16] = (_Float16)p1;
    }

    __builtin_amdgcn_wave_barrier();
    asm volatile("s_wait_dscnt 0x0" ::: "memory");

    const _Float16* prow = pb + l16 * 32;
    h16v pA = hcat(*(const h8v*)(prow + hb), *(const h8v*)(prow + hb + 16));

    // ---- O += P * V from LDS V ----
#pragma unroll
    for (int t = 0; t < 4; ++t) {
      const _Float16* vrow = &Vbuf[cur][(t * 16 + l16) * 32];
      h16v bV = hcat(*(const h8v*)(vrow + hb), *(const h8v*)(vrow + hb + 16));
      o[t] = __builtin_amdgcn_wmma_f32_16x16x32_f16(
          false, pA, false, bV, (short)0, o[t], false, false);
    }
    __syncthreads();   // protect buffer from next iteration's prefetch
  }

  // ---- epilogue ----
#pragma unroll
  for (int t = 0; t < 4; ++t) {
#pragma unroll
    for (int g = 0; g < 8; ++g) {
      const int r = q0 + g + hb;
      out[(size_t)r * HEAD + t * 16 + l16] = o[t][g] / lrow[g];
    }
  }
}

// ---------------------------------------------------------------------------
extern "C" void kernel_launch(void* const* d_in, const int* in_sizes, int n_in,
                              void* d_out, int out_size, void* d_ws, size_t ws_size,
                              hipStream_t stream) {
  const float* x  = (const float*)d_in[0];
  const float* Wq = (const float*)d_in[1];
  const float* bq = (const float*)d_in[2];
  const float* Wk = (const float*)d_in[3];
  const float* bk = (const float*)d_in[4];
  const float* Wv = (const float*)d_in[5];
  const float* bv = (const float*)d_in[6];
  float*       out = (float*)d_out;

  char* ws = (char*)d_ws;
  _Float16* WhT = (_Float16*)(ws);                                   // 384 KB
  _Float16* Qh  = (_Float16*)(ws + 384 * 1024);                      // 2 MB
  _Float16* Kh  = (_Float16*)(ws + 384 * 1024 + 2 * 1024 * 1024);    // 2 MB
  _Float16* VhT = (_Float16*)(ws + 384 * 1024 + 4 * 1024 * 1024);    // 2 MB

  wtrans_kernel<<<(3 * HEAD * D_MODEL) / 256, 256, 0, stream>>>(Wq, Wk, Wv, WhT);
  proj_kernel<<<NROWS / 64, 128, 0, stream>>>(x, WhT, bq, bk, bv, Qh, Kh, VhT);
  attn_kernel<<<NROWS / 64, 128, 0, stream>>>(Qh, Kh, VhT, out);
}
